// HAB_13013750907639
// MI455X (gfx1250) — compile-verified, matrix operand order checked by reading, not run on hardware
//
#include <hip/hip_runtime.h>
#include <hip/hip_bf16.h>
#include <math.h>

#define USE_ASYNC 1   // CDNA5 global_load_async_to_lds_* path (ASYNCcnt)

// ---------------------------------------------------------------------------
// Types for CDNA5 WMMA (wave32, v_wmma_f32_16x16x32_bf16)
// ---------------------------------------------------------------------------
typedef unsigned short u16;
typedef __attribute__((ext_vector_type(16))) __bf16 v16bf;
typedef __attribute__((ext_vector_type(8)))  float  v8f;
typedef __attribute__((ext_vector_type(4)))  unsigned int u32x4;

union Frag { v16bf v; u16 u[16]; u32x4 q[2]; };

__device__ __forceinline__ u16 f2bf(float f) {
  unsigned u = __float_as_uint(f);
  u += 0x7FFFu + ((u >> 16) & 1u);           // round-to-nearest-even
  return (u16)(u >> 16);
}
__device__ __forceinline__ v8f vzero8() {
  v8f z; for (int i = 0; i < 8; ++i) z[i] = 0.f; return z;
}

#define DIMC   180
#define HEADS  6
#define HD     30
#define HIMG   256
#define WIMG   256
#define MTOK   262144          // 4 * 256 * 256

// padded strides (zero-filled pads -> maskless b128 tile staging)
#define KP_X    192            // xn / attb / ln2 stride (180 -> 192)
#define KP_QKV  576            // qkv activations (540 -> 576)
#define KP_H1   768            // fc1 activations (720 -> 768)
#define KP_Y1   64             // conv mid activations (60 -> 64)

// ---------------------------------------------------------------------------
// Templated WMMA GEMM:  out[M x Nreal] = A[M x K] * Bw[Npad x K]^T (+ bias)
// AMODE 0: A linear, row stride CIN (== padded Ktot)
// AMODE 1: implicit-GEMM 3x3 conv gather, per-tap padded CIN, Ktot = 9*CIN
// EPI: 0 f32 | 1 bf16(+zero pad cols) | 2 gelu->bf16(+pad) |
//      3 proj residual fuse (f32) | 4 fc2 residual fuse (f32)
// ---------------------------------------------------------------------------
template<int AMODE, int CIN, int EPI>
__launch_bounds__(256)
__global__ void gemm_wmma(const u16* __restrict__ A, const u16* __restrict__ Bw,
                          const float* __restrict__ bias, void* __restrict__ outp,
                          int M, int Nreal, int Ktot, int ldo,
                          const float* __restrict__ p1,   // x (epi3) / res1 (epi4)
                          const float* __restrict__ p2,   // y2 (epi3)
                          const float* __restrict__ p3)   // sigmoid scale (epi3)
{
  __shared__ __align__(16) u16 As[2][128][48];   // 96B rows (16B multiple)
  __shared__ __align__(16) u16 Bs[2][64][48];

  const int tid  = threadIdx.x;
  const int lane = tid & 31;
  const int wv   = tid >> 5;          // 8 waves
  const int wm   = wv & 3;            // 4 waves x 32 rows
  const int wn   = wv >> 2;           // 2 waves x 32 cols
  const int m0   = blockIdx.x * 128;
  const int n0   = blockIdx.y * 64;
  const int khalf = lane >> 4, l16 = lane & 15;

  auto stage = [&](int buf, int k0) {
    // A tile: 128x32 u16 = 512 x b128 -> 2 per thread
#pragma unroll
    for (int it = 0; it < 2; ++it) {
      int e  = tid + it * 256;
      int mi = e >> 2, kk = (e & 3) * 8;
      int gm = m0 + mi, gk = k0 + kk;
      if (AMODE == 0) {
        const u16* src = A + (size_t)gm * CIN + gk;
#if USE_ASYNC
        unsigned lo = (unsigned)(size_t)(void*)&As[buf][mi][kk];
        asm volatile("global_load_async_to_lds_b128 %0, %1, off"
                     :: "v"(lo), "v"((unsigned long long)(size_t)src) : "memory");
#else
        *(u32x4*)&As[buf][mi][kk] = *(const u32x4*)src;
#endif
      } else {
        int tap = gk / CIN, ic = gk - tap * CIN;   // CIN const: cheap
        int b = gm >> 16, rem = gm & 65535;
        int y = rem >> 8, x = rem & 255;
        int yy = y + tap / 3 - 1, xx = x + tap % 3 - 1;
        u32x4 val; val[0] = val[1] = val[2] = val[3] = 0u;
        if (yy >= 0 && yy < HIMG && xx >= 0 && xx < WIMG)
          val = *(const u32x4*)(A + (size_t)((b << 16) + (yy << 8) + xx) * CIN + ic);
        *(u32x4*)&As[buf][mi][kk] = val;
      }
    }
    // B tile: 64x32 u16 = 256 x b128 -> 1 per thread
    {
      int ni = tid >> 2, kk = (tid & 3) * 8;
      const u16* src = Bw + (size_t)(n0 + ni) * Ktot + k0 + kk;
#if USE_ASYNC
      unsigned lo = (unsigned)(size_t)(void*)&Bs[buf][ni][kk];
      asm volatile("global_load_async_to_lds_b128 %0, %1, off"
                   :: "v"(lo), "v"((unsigned long long)(size_t)src) : "memory");
#else
      *(u32x4*)&Bs[buf][ni][kk] = *(const u32x4*)src;
#endif
    }
  };

  v8f acc[2][2];
  for (int i = 0; i < 2; ++i) for (int j = 0; j < 2; ++j) acc[i][j] = vzero8();

  stage(0, 0);
  int buf = 0;
  for (int k0 = 0; k0 < Ktot; k0 += 32) {
#if USE_ASYNC
    asm volatile("s_wait_asynccnt 0" ::: "memory");
#endif
    __syncthreads();                       // tile `buf` ready block-wide
    if (k0 + 32 < Ktot) stage(buf ^ 1, k0 + 32);   // prefetch next slice

    Frag af[2], bf[2];
#pragma unroll
    for (int mt = 0; mt < 2; ++mt) {
      const u16* ap = &As[buf][wm * 32 + mt * 16 + l16][khalf * 8];
      af[mt].q[0] = *(const u32x4*)ap;
      af[mt].q[1] = *(const u32x4*)(ap + 16);
    }
#pragma unroll
    for (int nt = 0; nt < 2; ++nt) {
      const u16* bp = &Bs[buf][wn * 32 + nt * 16 + l16][khalf * 8];
      bf[nt].q[0] = *(const u32x4*)bp;
      bf[nt].q[1] = *(const u32x4*)(bp + 16);
    }
#pragma unroll
    for (int mt = 0; mt < 2; ++mt)
#pragma unroll
      for (int nt = 0; nt < 2; ++nt)
        acc[mt][nt] = __builtin_amdgcn_wmma_f32_16x16x32_bf16(
            false, af[mt].v, false, bf[nt].v, (short)0, acc[mt][nt], false, false);
    buf ^= 1;
  }

  // epilogue: D layout n=lane%16, m=r+8*(lane/16)
#pragma unroll
  for (int mt = 0; mt < 2; ++mt)
#pragma unroll
    for (int nt = 0; nt < 2; ++nt)
#pragma unroll
      for (int r = 0; r < 8; ++r) {
        int row = m0 + wm * 32 + mt * 16 + r + khalf * 8;
        int col = n0 + wn * 32 + nt * 16 + l16;
        size_t o = (size_t)row * ldo + col;
        if (EPI == 1 || EPI == 2) {          // bf16 out, zero the pad columns
          u16 w = 0;
          if (col < Nreal) {
            float v = acc[mt][nt][r] + bias[col];
            if (EPI == 2) v = 0.5f * v * (1.f + erff(v * 0.70710678118f));
            w = f2bf(v);
          }
          ((u16*)outp)[o] = w;
        } else if (col < Nreal) {            // f32 outs are unpadded
          float v = acc[mt][nt][r] + bias[col];
          if (EPI == 0) {
            ((float*)outp)[o] = v;
          } else if (EPI == 3) {             // res1 = x + proj + 0.01*sig*y2
            int b = row >> 16;
            ((float*)outp)[o] = p1[o] + v + 0.01f * p3[b * DIMC + col] * p2[o];
          } else {                           // 4: out = res1 + fc2
            ((float*)outp)[o] = p1[o] + v;
          }
        }
      }
}

// ---------------------------------------------------------------------------
// Window attention: one block per (window, head), 128 threads = 4 waves.
// ---------------------------------------------------------------------------
__launch_bounds__(128)
__global__ void attn_wmma(const u16* __restrict__ qkv, const float* __restrict__ biasf,
                          u16* __restrict__ outa)
{
  __shared__ __align__(16) u16 Ks[64][32];   // [token][dim]
  __shared__ __align__(16) u16 Vt[32][64];   // [dim][token]
  __shared__ __align__(16) u16 Ps[64][64];   // probs bf16

  const int wid = blockIdx.x / HEADS;
  const int hh  = blockIdx.x % HEADS;
  const int b = wid >> 10, wrem = wid & 1023;
  const int wy = wrem >> 5, wx = wrem & 31;
  const int tid = threadIdx.x, lane = tid & 31, wv = tid >> 5;
  const int khalf = lane >> 4, l16 = lane & 15;

  auto rowOf = [&](int t) {
    int ty = t >> 3, tx = t & 7;
    return (b << 16) + ((wy * 8 + ty) << 8) + (wx * 8 + tx);
  };

  for (int e = tid; e < 64 * 32; e += 128) {
    int t = e >> 5, d = e & 31;
    size_t base = (size_t)rowOf(t) * KP_QKV + hh * HD;
    u16 kv = (d < HD) ? qkv[base + DIMC + d]     : (u16)0;
    u16 vv = (d < HD) ? qkv[base + 2 * DIMC + d] : (u16)0;
    Ks[t][d] = kv;
    Vt[d][t] = vv;
  }
  Frag qf;
  {
    int t = wv * 16 + l16;
    size_t base = (size_t)rowOf(t) * KP_QKV + hh * HD;
#pragma unroll
    for (int j = 0; j < 16; ++j) {
      int k = ((j >> 3) << 4) + khalf * 8 + (j & 7);
      qf.u[j] = (k < HD) ? qkv[base + k] : (u16)0;
    }
  }
  __syncthreads();

  v8f sacc[4];
#pragma unroll
  for (int nt = 0; nt < 4; ++nt) {
    Frag kf;
    const u16* bp = &Ks[nt * 16 + l16][khalf * 8];
    kf.q[0] = *(const u32x4*)bp;
    kf.q[1] = *(const u32x4*)(bp + 16);
    sacc[nt] = vzero8();
    sacc[nt] = __builtin_amdgcn_wmma_f32_16x16x32_bf16(
        false, qf.v, false, kf.v, (short)0, sacc[nt], false, false);
  }

  const float scale = 0.18257418583505536f;  // 1/sqrt(30)
  float sv[4][8];
#pragma unroll
  for (int nt = 0; nt < 4; ++nt)
#pragma unroll
    for (int r = 0; r < 8; ++r) {
      int rl = wv * 16 + r + khalf * 8;
      int n  = nt * 16 + l16;
      sv[nt][r] = sacc[nt][r] * scale + biasf[hh * 4096 + rl * 64 + n];
    }
#pragma unroll
  for (int r = 0; r < 8; ++r) {
    float mx = -1e30f;
#pragma unroll
    for (int nt = 0; nt < 4; ++nt) mx = fmaxf(mx, sv[nt][r]);
    for (int off = 8; off > 0; off >>= 1) mx = fmaxf(mx, __shfl_xor(mx, off, 16));
    float pv[4], sum = 0.f;
#pragma unroll
    for (int nt = 0; nt < 4; ++nt) { pv[nt] = __expf(sv[nt][r] - mx); sum += pv[nt]; }
    for (int off = 8; off > 0; off >>= 1) sum += __shfl_xor(sum, off, 16);
    float inv = 1.f / sum;
    int rl = wv * 16 + r + khalf * 8;
#pragma unroll
    for (int nt = 0; nt < 4; ++nt) Ps[rl][nt * 16 + l16] = f2bf(pv[nt] * inv);
  }
  __syncthreads();

  v8f oacc[2] = {vzero8(), vzero8()};
#pragma unroll
  for (int c = 0; c < 2; ++c) {
    Frag pf;
    const u16* pp = &Ps[wv * 16 + l16][c * 32 + khalf * 8];
    pf.q[0] = *(const u32x4*)pp;
    pf.q[1] = *(const u32x4*)(pp + 16);
#pragma unroll
    for (int nt = 0; nt < 2; ++nt) {
      Frag vf;
      const u16* vp = &Vt[nt * 16 + l16][c * 32 + khalf * 8];
      vf.q[0] = *(const u32x4*)vp;
      vf.q[1] = *(const u32x4*)(vp + 16);
      oacc[nt] = __builtin_amdgcn_wmma_f32_16x16x32_bf16(
          false, pf.v, false, vf.v, (short)0, oacc[nt], false, false);
    }
  }
#pragma unroll
  for (int nt = 0; nt < 2; ++nt)
#pragma unroll
    for (int r = 0; r < 8; ++r) {
      int rl = wv * 16 + r + khalf * 8;
      int d  = nt * 16 + l16;
      if (d < HD)
        outa[(size_t)rowOf(rl) * KP_X + hh * HD + d] = f2bf(oacc[nt][r]);
    }
  // zero attb pad columns 180..191 (duplicated across heads; same zeros)
  for (int e = tid; e < 64 * (KP_X - DIMC); e += 128) {
    int rl = e / (KP_X - DIMC), c = e % (KP_X - DIMC);
    outa[(size_t)rowOf(rl) * KP_X + DIMC + c] = 0;
  }
}

// ---------------------------------------------------------------------------
// LayerNorm (C=180) -> bf16 with zero-padded stride 192. Wave per token.
// ---------------------------------------------------------------------------
__launch_bounds__(256)
__global__ void ln_bf16(const float* __restrict__ x, const float* __restrict__ g,
                        const float* __restrict__ bb, u16* __restrict__ out, int Mtok)
{
  int wv = threadIdx.x >> 5, lane = threadIdx.x & 31;
  int tok = blockIdx.x * 8 + wv;
  if (tok >= Mtok) return;
  const float* xp = x + (size_t)tok * DIMC;
  float vv[6], s = 0.f, s2 = 0.f;
#pragma unroll
  for (int i = 0; i < 6; ++i) {
    int c = lane + i * 32;
    float v = (c < DIMC) ? xp[c] : 0.f;
    vv[i] = v; s += v; s2 += v * v;
  }
  for (int off = 16; off > 0; off >>= 1) {
    s  += __shfl_xor(s,  off, 32);
    s2 += __shfl_xor(s2, off, 32);
  }
  float mean = s * (1.f / DIMC);
  float var  = s2 * (1.f / DIMC) - mean * mean;
  float rstd = rsqrtf(var + 1e-5f);
  u16* op = out + (size_t)tok * KP_X;
#pragma unroll
  for (int i = 0; i < 6; ++i) {                 // 6*32 == 192 == KP_X exactly
    int c = lane + i * 32;
    op[c] = (c < DIMC) ? f2bf((vv[i] - mean) * rstd * g[c] + bb[c]) : (u16)0;
  }
}

// ---------------------------------------------------------------------------
// Channel-attention helpers
// ---------------------------------------------------------------------------
__global__ void pool_kernel(const float* __restrict__ y2, float* __restrict__ pool)
{
  int bc = blockIdx.x, b = bc / DIMC, c = bc % DIMC;
  __shared__ float red[256];
  float s = 0.f;
  for (int i = threadIdx.x; i < 65536; i += 256)
    s += y2[((size_t)(b << 16) + i) * DIMC + c];
  red[threadIdx.x] = s;
  __syncthreads();
  for (int o = 128; o > 0; o >>= 1) {
    if (threadIdx.x < o) red[threadIdx.x] += red[threadIdx.x + o];
    __syncthreads();
  }
  if (threadIdx.x == 0) pool[bc] = red[0] * (1.f / 65536.f);
}

__global__ void ca_kernel(const float* __restrict__ pool,
                          const float* __restrict__ w1, const float* __restrict__ b1,
                          const float* __restrict__ w2, const float* __restrict__ b2,
                          float* __restrict__ sig)
{
  int b = blockIdx.x;
  __shared__ float mid[6];
  if (threadIdx.x < 6) {
    float s = b1[threadIdx.x];
    for (int c = 0; c < DIMC; ++c) s += w1[threadIdx.x * DIMC + c] * pool[b * DIMC + c];
    mid[threadIdx.x] = fmaxf(s, 0.f);
  }
  __syncthreads();
  for (int c = threadIdx.x; c < DIMC; c += 256) {
    float s = b2[c];
    for (int j = 0; j < 6; ++j) s += w2[c * 6 + j] * mid[j];
    sig[b * DIMC + c] = 1.f / (1.f + __expf(-s));
  }
}

// ---------------------------------------------------------------------------
// Weight packing (padded, zero-filled) / bias expansion
// ---------------------------------------------------------------------------
__global__ void pack_linear(const float* __restrict__ in, u16* __restrict__ out,
                            int Nr, int Kr, int NP, int KP)
{ // out[NP][KP], zero pads
  int i = blockIdx.x * 256 + threadIdx.x;
  if (i >= NP * KP) return;
  int n = i / KP, k = i - n * KP;
  out[i] = (n < Nr && k < Kr) ? f2bf(in[n * Kr + k]) : (u16)0;
}

__global__ void pack_conv(const float* __restrict__ in, u16* __restrict__ out,
                          int Cout, int Cin, int CINP, int NP)
{ // OIHW -> [oc][tap][ic] bf16, per-tap padded CIN
  int n = NP * 9 * CINP;
  int i = blockIdx.x * 256 + threadIdx.x;
  if (i >= n) return;
  int oc = i / (9 * CINP);
  int r  = i - oc * 9 * CINP;
  int tap = r / CINP, ic = r - tap * CINP;
  out[i] = (oc < Cout && ic < Cin) ? f2bf(in[(oc * Cin + ic) * 9 + tap]) : (u16)0;
}

__global__ void bias_expand(const float* __restrict__ tbl, const int* __restrict__ rpi,
                            float* __restrict__ bf)
{
  int i = blockIdx.x * 256 + threadIdx.x;
  if (i >= HEADS * 4096) return;
  int hh = i >> 12, ij = i & 4095;
  bf[i] = tbl[rpi[ij] * HEADS + hh];
}

// ---------------------------------------------------------------------------
// Orchestration
// ---------------------------------------------------------------------------
extern "C" void kernel_launch(void* const* d_in, const int* in_sizes, int n_in,
                              void* d_out, int out_size, void* d_ws, size_t ws_size,
                              hipStream_t stream)
{
  (void)in_sizes; (void)n_in; (void)out_size; (void)ws_size;
  const float* x         = (const float*)d_in[0];
  const float* ln1_g     = (const float*)d_in[1];
  const float* ln1_b     = (const float*)d_in[2];
  const float* qkv_w     = (const float*)d_in[3];
  const float* qkv_b     = (const float*)d_in[4];
  const float* bias_tab  = (const float*)d_in[5];
  const float* proj_w    = (const float*)d_in[6];
  const float* proj_b    = (const float*)d_in[7];
  const float* cab_w1    = (const float*)d_in[8];
  const float* cab_b1    = (const float*)d_in[9];
  const float* cab_w2    = (const float*)d_in[10];
  const float* cab_b2    = (const float*)d_in[11];
  const float* ca_w1     = (const float*)d_in[12];
  const float* ca_b1     = (const float*)d_in[13];
  const float* ca_w2     = (const float*)d_in[14];
  const float* ca_b2     = (const float*)d_in[15];
  const float* ln2_g     = (const float*)d_in[16];
  const float* ln2_b     = (const float*)d_in[17];
  const float* fc1_w     = (const float*)d_in[18];
  const float* fc1_b     = (const float*)d_in[19];
  const float* fc2_w     = (const float*)d_in[20];
  const float* fc2_b     = (const float*)d_in[21];
  const int*   rpi       = (const int*)d_in[22];

  char* ws = (char*)d_ws;
  size_t off = 0;
  auto alloc = [&](size_t bytes) -> void* {
    void* p = (void*)(ws + off);
    off += (bytes + 255) & ~(size_t)255;
    return p;
  };
  u16*   xn    = (u16*)  alloc((size_t)MTOK * KP_X   * 2);
  u16*   qkvb  = (u16*)  alloc((size_t)MTOK * KP_QKV * 2);
  u16*   attb  = (u16*)  alloc((size_t)MTOK * KP_X   * 2);
  u16*   y1    = (u16*)  alloc((size_t)MTOK * KP_Y1  * 2);
  float* y2    = (float*)alloc((size_t)MTOK * DIMC * 4);
  float* res1  = (float*)alloc((size_t)MTOK * DIMC * 4);
  u16*   ln2bf = (u16*)  alloc((size_t)MTOK * KP_X   * 2);
  u16*   h1    = (u16*)  alloc((size_t)MTOK * KP_H1  * 2);
  float* pool  = (float*)alloc(720 * 4);
  float* sig   = (float*)alloc(720 * 4);
  float* biasf = (float*)alloc(HEADS * 4096 * 4);
  u16*   wq    = (u16*)  alloc((size_t)576 * 192 * 2);
  u16*   wp    = (u16*)  alloc((size_t)192 * 192 * 2);
  u16*   w1p   = (u16*)  alloc((size_t)768 * 192 * 2);
  u16*   w2p   = (u16*)  alloc((size_t)192 * 768 * 2);
  u16*   wc1   = (u16*)  alloc((size_t)64  * 1728 * 2);
  u16*   wc2   = (u16*)  alloc((size_t)192 * 576 * 2);

  // weight packing (bf16, zero-padded)
  pack_linear<<<(576*192 + 255)/256, 256, 0, stream>>>(qkv_w, wq, 540, 180, 576, 192);
  pack_linear<<<(192*192 + 255)/256, 256, 0, stream>>>(proj_w, wp, 180, 180, 192, 192);
  pack_linear<<<(768*192 + 255)/256, 256, 0, stream>>>(fc1_w, w1p, 720, 180, 768, 192);
  pack_linear<<<(192*768 + 255)/256, 256, 0, stream>>>(fc2_w, w2p, 180, 720, 192, 768);
  pack_conv<<<(64*1728 + 255)/256, 256, 0, stream>>>(cab_w1, wc1, 60, 180, 192, 64);
  pack_conv<<<(192*576 + 255)/256, 256, 0, stream>>>(cab_w2, wc2, 180, 60, 64, 192);
  bias_expand<<<(HEADS*4096)/256, 256, 0, stream>>>(bias_tab, rpi, biasf);

  // LN1 -> xn (bf16, stride 192)
  ln_bf16<<<MTOK/8, 256, 0, stream>>>(x, ln1_g, ln1_b, xn, MTOK);

  // qkv = xn @ qkv_w^T + b  (bf16, stride 576)
  gemm_wmma<0, KP_X, 1><<<dim3(MTOK/128, 9), 256, 0, stream>>>(
      xn, wq, qkv_b, qkvb, MTOK, 540, 192, KP_QKV, nullptr, nullptr, nullptr);

  // window attention
  attn_wmma<<<4096 * HEADS, 128, 0, stream>>>(qkvb, biasf, attb);

  // CAB conv1 (implicit GEMM + GELU) -> y1 (bf16, stride 64)
  gemm_wmma<1, KP_X, 2><<<dim3(MTOK/128, 1), 256, 0, stream>>>(
      xn, wc1, cab_b1, y1, MTOK, 60, 9 * KP_X, KP_Y1, nullptr, nullptr, nullptr);
  // CAB conv2 (implicit GEMM) -> y2 (f32, stride 180)
  gemm_wmma<1, KP_Y1, 0><<<dim3(MTOK/128, 3), 256, 0, stream>>>(
      y1, wc2, cab_b2, y2, MTOK, 180, 9 * KP_Y1, DIMC, nullptr, nullptr, nullptr);
  // channel attention
  pool_kernel<<<720, 256, 0, stream>>>(y2, pool);
  ca_kernel<<<4, 256, 0, stream>>>(pool, ca_w1, ca_b1, ca_w2, ca_b2, sig);

  // proj + fused residual: res1 = x + (attn @ proj_w^T + b) + 0.01*sig*y2
  gemm_wmma<0, KP_X, 3><<<dim3(MTOK/128, 3), 256, 0, stream>>>(
      attb, wp, proj_b, res1, MTOK, 180, 192, DIMC, x, y2, sig);

  // LN2 -> bf16 (stride 192)
  ln_bf16<<<MTOK/8, 256, 0, stream>>>(res1, ln2_g, ln2_b, ln2bf, MTOK);

  // fc1 + GELU -> h1 (bf16, stride 768)
  gemm_wmma<0, KP_X, 2><<<dim3(MTOK/128, 12), 256, 0, stream>>>(
      ln2bf, w1p, fc1_b, h1, MTOK, 720, 192, KP_H1, nullptr, nullptr, nullptr);

  // fc2 + residual -> d_out (f32)
  gemm_wmma<0, KP_H1, 4><<<dim3(MTOK/128, 3), 256, 0, stream>>>(
      h1, w2p, fc2_b, (float*)d_out, MTOK, 180, KP_H1, DIMC, res1, nullptr, nullptr);
}
